// QDKTNet_23270132809817
// MI455X (gfx1250) — matrix operand order repaired and checked by reading, not compiled, not run on hardware
//
#include <hip/hip_runtime.h>
#include <hip/hip_bf16.h>

typedef __attribute__((ext_vector_type(16))) _Float16 v16h;
typedef __attribute__((ext_vector_type(8)))  float    v8f;
typedef __attribute__((ext_vector_type(4)))  float    f32x4;
typedef __attribute__((ext_vector_type(4)))  int      v4i;

#define Bn   64
#define Tn   200
#define Ln   201
#define En   512
#define Hn   512
#define Gn   2048
#define NQn  10000
#define NWG  32        // persistent LSTM workgroups; each owns Hn/NWG = 16 hidden units
#define BROW 520       // padded LDS row stride (halves): 520*2B=1040B -> stride 260 dwords = 4 mod 64 banks

union FragU { v16h v; f32x4 q[2]; };

// A fragment (16x32 f16, MxK): lane row = lane&15, half = lane>>4.
__device__ __forceinline__ v16h load_a_frag(const _Float16* rowp, int half) {
  FragU u;
  u.q[0] = *(const f32x4*)(rowp + 16 * half);       // K: 16h .. 16h+7
  u.q[1] = *(const f32x4*)(rowp + 16 * half + 8);   // K: 16h+8 .. 16h+15
  return u.v;
}

// B fragment (32x16 f16, KxN) from row-major W[n][k] (= B^T): 16 contiguous K from kb+16*half.
__device__ __forceinline__ v16h load_b_frag(const _Float16* rowp, int half) {
  FragU u;
  u.q[0] = *(const f32x4*)(rowp + 16 * half);       // K: 16h .. 16h+7
  u.q[1] = *(const f32x4*)(rowp + 16 * half + 8);   // K: 16h+8 .. 16h+15
  return u.v;
}

__device__ __forceinline__ v8f wmma_f16(v16h a, v16h b, v8f c) {
  return __builtin_amdgcn_wmma_f32_16x16x32_f16(false, a, false, b, (short)0, c, false, false);
}

__device__ __forceinline__ float sigf(float x) { return 1.0f / (1.0f + expf(-x)); }

// ---------------- kernel 1: embedding gather + f32->f16 ----------------
__global__ void __launch_bounds__(256) k_gather(const int* __restrict__ q,
                                                const int* __restrict__ r,
                                                const float* __restrict__ emb,
                                                _Float16* __restrict__ Xe) {
  const int row = blockIdx.x;                 // b*Tn + t
  const int b = row / Tn, t = row % Tn;
  const int idx = q[b * Ln + t] + NQn * r[b * Ln + t];
  const float* __restrict__ src = emb + (size_t)idx * En;
  _Float16* __restrict__ dst = Xe + (size_t)row * En;
  for (int e = threadIdx.x; e < En; e += 256) dst[e] = (_Float16)src[e];
}

// ------------- kernel 2: weight conversion + state/barrier init -------------
__global__ void __launch_bounds__(256) k_convert_init(const float* __restrict__ Wih,
                                                      const float* __restrict__ Whh,
                                                      _Float16* __restrict__ WihH,
                                                      _Float16* __restrict__ WhhH,
                                                      _Float16* __restrict__ h16,
                                                      float* __restrict__ cst,
                                                      unsigned* __restrict__ bar) {
  const size_t i = (size_t)blockIdx.x * 256 + threadIdx.x;   // grid covers Gn*En = 1,048,576
  WihH[i] = (_Float16)Wih[i];
  WhhH[i] = (_Float16)Whh[i];
  if (i < 2u * Bn * Hn) h16[i] = (_Float16)0.0f;             // both ping-pong h buffers
  if (i < (size_t)Bn * Hn) cst[i] = 0.0f;
  if (i < 8) bar[i] = 0u;
}

// ---------------- kernel 3: xi = Xe @ W_ih^T + (b_ih + b_hh) ----------------
// Block = 8 waves -> 32x256 tile. Wave = 16x64 (4 wmma accumulators).
// Branch-free software pipeline: steady loop loads kb+32 unconditionally
// (trip count En/32-1), last k-block peeled; full unroll renames the
// rotating fragment registers away.
__global__ void __launch_bounds__(256) k_gemm_xi(const _Float16* __restrict__ Xe,
                                                 const _Float16* __restrict__ WihH,
                                                 const float* __restrict__ b_ih,
                                                 const float* __restrict__ b_hh,
                                                 float* __restrict__ Xi) {
  const int w = threadIdx.x >> 5, l = threadIdx.x & 31;
  const int col = l & 15, half = l >> 4;
  const int mblk = blockIdx.x >> 3;                 // 0..399
  const int nblk = blockIdx.x & 7;                  // 0..7
  const int m0 = mblk * 32 + (w >> 2) * 16;
  const int n0 = nblk * 256 + (w & 3) * 64;

  v8f acc[4];
  #pragma unroll
  for (int i = 0; i < 4; ++i) {
    const float bv = b_ih[n0 + i * 16 + col] + b_hh[n0 + i * 16 + col];
    #pragma unroll
    for (int e = 0; e < 8; ++e) acc[i][e] = bv;
  }

  const _Float16* __restrict__ arow = Xe + (size_t)(m0 + col) * En;
  const _Float16* __restrict__ brow[4];
  #pragma unroll
  for (int i = 0; i < 4; ++i) brow[i] = WihH + (size_t)(n0 + i * 16 + col) * En;

  v16h a0 = load_a_frag(arow, half);
  v16h b0[4];
  #pragma unroll
  for (int i = 0; i < 4; ++i) b0[i] = load_b_frag(brow[i], half);

  #pragma unroll
  for (int kb = 0; kb < En - 32; kb += 32) {
    const v16h a1 = load_a_frag(arow + kb + 32, half);
    v16h b1[4];
    #pragma unroll
    for (int i = 0; i < 4; ++i) b1[i] = load_b_frag(brow[i] + kb + 32, half);
    #pragma unroll
    for (int i = 0; i < 4; ++i) acc[i] = wmma_f16(a0, b0[i], acc[i]);
    a0 = a1;
    #pragma unroll
    for (int i = 0; i < 4; ++i) b0[i] = b1[i];
  }
  #pragma unroll
  for (int i = 0; i < 4; ++i) acc[i] = wmma_f16(a0, b0[i], acc[i]);   // peeled last k-block

  #pragma unroll
  for (int i = 0; i < 4; ++i)
    #pragma unroll
    for (int v = 0; v < 8; ++v)
      Xi[(size_t)(m0 + v + 8 * half) * Gn + n0 + i * 16 + col] = acc[i][v];
}

// ---------------- device-wide barrier (generation counter) ------------------
__device__ __forceinline__ void grid_barrier(unsigned* __restrict__ bar, unsigned nwg) {
  __threadfence();
  __syncthreads();
  if (threadIdx.x == 0) {
    const unsigned gen = __hip_atomic_load(bar + 1, __ATOMIC_ACQUIRE, __HIP_MEMORY_SCOPE_AGENT);
    const unsigned arr = __hip_atomic_fetch_add(bar, 1u, __ATOMIC_ACQ_REL, __HIP_MEMORY_SCOPE_AGENT);
    if (arr == nwg - 1u) {
      __hip_atomic_store(bar, 0u, __ATOMIC_RELAXED, __HIP_MEMORY_SCOPE_AGENT);
      __hip_atomic_fetch_add(bar + 1, 1u, __ATOMIC_ACQ_REL, __HIP_MEMORY_SCOPE_AGENT);
    } else {
      while (__hip_atomic_load(bar + 1, __ATOMIC_ACQUIRE, __HIP_MEMORY_SCOPE_AGENT) == gen)
        __builtin_amdgcn_s_sleep(2);
    }
  }
  __syncthreads();
}

// ---------------- kernel 4: persistent LSTM recurrence ----------------------
// WG `wg` owns hidden slice [j0, j0+16). Its W_hh slice (64 gate-rows x 512 K,
// 64 KB f16) is time-invariant -> staged in LDS once (async-to-LDS on the
// gfx1250 path), then read 200x via ds_load_b128.
// Dynamic LDS: bsh = 64 rows x BROW halves (66560 B), gates = 64x64 f32 (16384 B).
__global__ void __launch_bounds__(256) k_lstm(const float* __restrict__ Xi,
                                              const _Float16* __restrict__ WhhH,
                                              _Float16* __restrict__ h16,   // 2 * Bn*Hn (ping-pong)
                                              float* __restrict__ cst,
                                              float* __restrict__ Hall,
                                              unsigned* __restrict__ bar) {
  extern __shared__ char smem[];
  _Float16* bsh = (_Float16*)smem;                        // 64 * BROW halves
  float* gates = (float*)(smem + 64 * BROW * 2);          // Bn * 64 floats

  const int wg = blockIdx.x;                       // 0..NWG-1
  const int j0 = wg * (Hn / NWG);                  // 16 hidden units per WG
  const int w = threadIdx.x >> 5, l = threadIdx.x & 31;
  const int col = l & 15, half = l >> 4;
  const int mt = w >> 1;                           // 0..3 (batch tile)
  const int nh = w & 1;                            // 0..1 -> gates {i,f} or {g,o}

  // ---- stage W_hh slice into LDS: local row gc in [0,64) holds global row
  // (gc>>4)*Hn + j0 + (gc&15); 64 chunks of 16B per row ----
  for (int c = threadIdx.x; c < 64 * (En / 8); c += 256) {
    const int gc = c >> 6;                         // local gate-column row
    const int kc = c & 63;                         // 16B chunk within row
    const _Float16* src = WhhH + (size_t)((gc >> 4) * Hn + j0 + (gc & 15)) * Hn + kc * 8;
    _Float16* dst = bsh + gc * BROW + kc * 8;
#if __has_builtin(__builtin_amdgcn_global_load_async_to_lds_b128)
    v4i* gsrc = (v4i*)(void*)src;                  // strip const, reinterpret
    v4i* ldst = (v4i*)(void*)dst;
    __builtin_amdgcn_global_load_async_to_lds_b128(
        (__attribute__((address_space(1))) v4i*)gsrc,
        (__attribute__((address_space(3))) v4i*)ldst, 0, 0);
#else
    *(f32x4*)dst = *(const f32x4*)src;
#endif
  }
#if __has_builtin(__builtin_amdgcn_global_load_async_to_lds_b128)
#if __has_builtin(__builtin_amdgcn_s_wait_asynccnt)
  __builtin_amdgcn_s_wait_asynccnt(0);
#else
  asm volatile("s_wait_asynccnt 0" ::: "memory");
#endif
#endif
  __syncthreads();

  const _Float16* bl0 = bsh + (size_t)((nh * 2 + 0) * 16 + col) * BROW;
  const _Float16* bl1 = bsh + (size_t)((nh * 2 + 1) * 16 + col) * BROW;

  for (int t = 0; t < Tn; ++t) {
    const _Float16* __restrict__ hprev = h16 + (size_t)(t & 1) * Bn * Hn;
    _Float16* __restrict__ hnext = h16 + (size_t)((t + 1) & 1) * Bn * Hn;

    v8f acc[2];
    #pragma unroll
    for (int i = 0; i < 2; ++i)
      #pragma unroll
      for (int e = 0; e < 8; ++e) acc[i][e] = 0.0f;

    const _Float16* __restrict__ arow = hprev + (size_t)(mt * 16 + col) * Hn;
    v16h a0 = load_a_frag(arow, half);             // A from global (L2), double-buffered
    #pragma unroll
    for (int kb = 0; kb < Hn - 32; kb += 32) {
      const v16h a1 = load_a_frag(arow + kb + 32, half);
      const v16h b0 = load_b_frag(bl0 + kb, half); // B from LDS -> ds_load_b128
      acc[0] = wmma_f16(a0, b0, acc[0]);
      const v16h b1 = load_b_frag(bl1 + kb, half);
      acc[1] = wmma_f16(a0, b1, acc[1]);
      a0 = a1;
    }
    {
      const v16h b0 = load_b_frag(bl0 + Hn - 32, half);   // peeled last k-block
      acc[0] = wmma_f16(a0, b0, acc[0]);
      const v16h b1 = load_b_frag(bl1 + Hn - 32, half);
      acc[1] = wmma_f16(a0, b1, acc[1]);
    }

    #pragma unroll
    for (int i = 0; i < 2; ++i)
      #pragma unroll
      for (int v = 0; v < 8; ++v)
        gates[(mt * 16 + v + 8 * half) * 64 + (nh * 2 + i) * 16 + col] = acc[i][v];
    __syncthreads();

    // elementwise gates for this WG's 64x16 hidden slice
    #pragma unroll
    for (int e = 0; e < 4; ++e) {
      const int p = threadIdx.x + e * 256;         // 0..1023
      const int m = p >> 4, j = p & 15;
      const size_t xibase = ((size_t)m * Tn + t) * Gn + j0 + j;
      float gi = gates[m * 64 +  0 + j] + Xi[xibase + 0 * Hn];
      float gf = gates[m * 64 + 16 + j] + Xi[xibase + 1 * Hn];
      float gg = gates[m * 64 + 32 + j] + Xi[xibase + 2 * Hn];
      float go = gates[m * 64 + 48 + j] + Xi[xibase + 3 * Hn];
      gi = sigf(gi); gf = sigf(gf); gg = tanhf(gg); go = sigf(go);
      const int ci = m * Hn + j0 + j;
      const float c = gf * cst[ci] + gi * gg;
      cst[ci] = c;
      const float h = go * tanhf(c);
      hnext[ci] = (_Float16)h;
      Hall[((size_t)m * Tn + t) * Hn + j0 + j] = h;
    }
    __syncthreads();
    grid_barrier(bar, NWG);
  }
}

// ---------------- kernel 5: out[b,t] = sigmoid(h . W_out[qshft] + b_out) ----
__global__ void __launch_bounds__(256) k_out(const float* __restrict__ Hall,
                                             const float* __restrict__ W_out,
                                             const float* __restrict__ b_out,
                                             const int* __restrict__ qshft,
                                             float* __restrict__ out) {
  const int gid = blockIdx.x * 8 + (threadIdx.x >> 5);   // (b*Tn + t), one wave each
  const int l = threadIdx.x & 31;
  const int b = gid / Tn, t = gid % Tn;
  const int idx = qshft[b * Tn + t];
  const float* __restrict__ h  = Hall + (size_t)gid * Hn;
  const float* __restrict__ wo = W_out + (size_t)idx * Hn;
  float s = 0.0f;
  for (int k = l; k < Hn; k += 32) s += h[k] * wo[k];
  #pragma unroll
  for (int off = 16; off > 0; off >>= 1) s += __shfl_down(s, off, 32);
  if (l == 0) out[gid] = 1.0f / (1.0f + expf(-(s + b_out[idx])));
}

extern "C" void kernel_launch(void* const* d_in, const int* in_sizes, int n_in,
                              void* d_out, int out_size, void* d_ws, size_t ws_size,
                              hipStream_t stream) {
  (void)in_sizes; (void)n_in; (void)out_size; (void)ws_size;
  const int*   q     = (const int*)  d_in[0];
  // d_in[1] = c (unused by reference output path)
  const int*   r     = (const int*)  d_in[2];
  const int*   qshft = (const int*)  d_in[3];
  const float* emb   = (const float*)d_in[4];
  const float* W_ih  = (const float*)d_in[5];
  const float* W_hh  = (const float*)d_in[6];
  const float* b_ih  = (const float*)d_in[7];
  const float* b_hh  = (const float*)d_in[8];
  const float* W_out = (const float*)d_in[9];
  const float* b_out = (const float*)d_in[10];
  float* out = (float*)d_out;

  char* ws = (char*)d_ws;
  size_t off = 0;
  auto alloc = [&](size_t bytes) -> void* {
    void* p = ws + off;
    off = (off + bytes + 255) & ~(size_t)255;
    return p;
  };
  _Float16* Xe   = (_Float16*)alloc((size_t)Bn * Tn * En * 2);   // 13.1 MB
  _Float16* WihH = (_Float16*)alloc((size_t)Gn * En * 2);        //  2.1 MB
  _Float16* WhhH = (_Float16*)alloc((size_t)Gn * Hn * 2);        //  2.1 MB
  float*    Xi   = (float*)   alloc((size_t)Bn * Tn * Gn * 4);   // 104.9 MB
  float*    Hall = (float*)   alloc((size_t)Bn * Tn * Hn * 4);   // 26.2 MB
  _Float16* h16  = (_Float16*)alloc((size_t)2 * Bn * Hn * 2);    // ping-pong h
  float*    cst  = (float*)   alloc((size_t)Bn * Hn * 4);
  unsigned* bar  = (unsigned*)alloc(256);

  const size_t lstm_lds = (size_t)64 * BROW * 2 + (size_t)Bn * 64 * 4;  // 82944 B (<320KB/WG on CDNA5)

  k_gather<<<Bn * Tn, 256, 0, stream>>>(q, r, emb, Xe);
  k_convert_init<<<(Gn * En) / 256, 256, 0, stream>>>(W_ih, W_hh, WihH, WhhH, h16, cst, bar);
  k_gemm_xi<<<(Bn * Tn / 32) * (Gn / 256), 256, 0, stream>>>(Xe, WihH, b_ih, b_hh, Xi);
  k_lstm<<<NWG, 256, lstm_lds, stream>>>(Xi, WhhH, h16, cst, Hall, bar);
  k_out<<<(Bn * Tn) / 8, 256, 0, stream>>>(Hall, W_out, b_out, qshft, out);
}
